// SystemRNN_41575283425535
// MI455X (gfx1250) — matrix operand order repaired and checked by reading, not compiled, not run on hardware
//
#include <hip/hip_runtime.h>

#define T_LEN 8192
#define IN_SZ 512
#define HID 2048
#define NWG 64
#define NTHR (NWG * 256)
#define NT 4              // 16x16 N-tiles per wave in the GEMM

typedef __attribute__((ext_vector_type(16))) __bf16 v16bf;
typedef __attribute__((ext_vector_type(8)))  float  v8f;
typedef __attribute__((ext_vector_type(4)))  unsigned u32x4;
typedef __attribute__((ext_vector_type(8)))  unsigned u32x8;

union ABFrag { uint4 u[2]; v16bf v; };   // 32 bytes = one wave-lane WMMA fragment

// ---------------------------------------------------------------------------
// Pre-pass A: xT_bf16[T, IN] = transpose(x[IN, T]) with f32->bf16 convert.
// ---------------------------------------------------------------------------
__global__ void __launch_bounds__(256) transpose_conv_kernel(
    const float* __restrict__ x, __bf16* __restrict__ xbf)
{
    __shared__ __bf16 tile[32][33];
    const int tx = threadIdx.x & 31;
    const int ty = threadIdx.x >> 5;          // 32x8 threads
    const int t0 = blockIdx.x * 32;           // 256 blocks along T
    const int i0 = blockIdx.y * 32;           // 16 blocks along IN
#pragma unroll
    for (int r = ty; r < 32; r += 8)
        tile[r][tx] = (__bf16)x[(size_t)(i0 + r) * T_LEN + t0 + tx];
    __syncthreads();
#pragma unroll
    for (int r = ty; r < 32; r += 8)
        xbf[(size_t)(t0 + r) * IN_SZ + i0 + tx] = tile[tx][r];
}

// ---------------------------------------------------------------------------
// Pre-pass B: Wx f32 -> bf16 (same [IN, HID] layout).
// ---------------------------------------------------------------------------
__global__ void __launch_bounds__(256) convw_kernel(
    const float* __restrict__ w, __bf16* __restrict__ wb)
{
    int i = blockIdx.x * 256 + threadIdx.x;
    wb[i] = (__bf16)w[i];
}

// ---------------------------------------------------------------------------
// Kernel 1: hxall[T, H] = xT[T, IN] @ Wx[IN, H] via bf16 WMMA, f32 accum.
// One wave computes a 16x64 strip (4 WMMA tiles sharing the A fragment).
// ---------------------------------------------------------------------------
__global__ void __launch_bounds__(256) gemm_hx_kernel(
    const __bf16* __restrict__ xbf,   // [T_LEN, IN_SZ]
    const __bf16* __restrict__ wxbf,  // [IN_SZ, HID]
    float* __restrict__ hxall)        // [T_LEN, HID]
{
    const int wave = threadIdx.x >> 5;
    const int lane = threadIdx.x & 31;
    const int gw = blockIdx.x * 8 + wave;        // 512 * 32 wave-strips
    const int nGroups = HID / (16 * NT);         // 32
    const int tileM = gw / nGroups;
    const int n0 = (gw % nGroups) * (16 * NT);
    const int t0 = tileM * 16;
    const int lm = lane & 15;
    const int lh = lane >> 4;

    v8f c[NT] = {};
    for (int k0 = 0; k0 < IN_SZ; k0 += 32) {
        ABFrag af;
        const __bf16* arow = xbf + (size_t)(t0 + lm) * IN_SZ + k0 + 8 * lh;
        af.u[0] = *(const uint4*)(arow);
        af.u[1] = *(const uint4*)(arow + 16);
        const __bf16* brow = wxbf + (size_t)(k0 + 8 * lh) * HID + n0 + lm;
#pragma unroll
        for (int n = 0; n < NT; ++n) {
            ABFrag bf;
#pragma unroll
            for (int s = 0; s < 16; ++s)
                bf.v[s] = brow[(size_t)((s & 7) + 16 * (s >> 3)) * HID + n * 16];
            c[n] = __builtin_amdgcn_wmma_f32_16x16x32_bf16(
                       false, af.v, false, bf.v, (short)0, c[n], false, false);
        }
    }
    // C/D layout: element r -> M = r + 8*lh, N = lm
#pragma unroll
    for (int n = 0; n < NT; ++n)
#pragma unroll
        for (int r = 0; r < 8; ++r)
            hxall[(size_t)(t0 + r + 8 * lh) * HID + n0 + n * 16 + lm] = c[n][r];
}

// ---------------------------------------------------------------------------
// Grid-wide software barrier (64 resident workgroups, generation counter).
// ---------------------------------------------------------------------------
__device__ __forceinline__ void grid_barrier(unsigned* arrive, unsigned* gen)
{
    __syncthreads();
    if (threadIdx.x == 0) {
        __threadfence();
        unsigned g = __hip_atomic_load(gen, __ATOMIC_ACQUIRE, __HIP_MEMORY_SCOPE_AGENT);
        unsigned old = __hip_atomic_fetch_add(arrive, 1u, __ATOMIC_ACQ_REL, __HIP_MEMORY_SCOPE_AGENT);
        if (old == NWG - 1) {
            __hip_atomic_store(arrive, 0u, __ATOMIC_RELAXED, __HIP_MEMORY_SCOPE_AGENT);
            __hip_atomic_fetch_add(gen, 1u, __ATOMIC_RELEASE, __HIP_MEMORY_SCOPE_AGENT);
        } else {
            while (__hip_atomic_load(gen, __ATOMIC_ACQUIRE, __HIP_MEMORY_SCOPE_AGENT) == g)
                __builtin_amdgcn_s_sleep(2);
        }
        __threadfence();
    }
    __syncthreads();
}

// ---------------------------------------------------------------------------
// Kernel 2: persistent scan. Each WG keeps its 256x256 f32 slab of Wh in LDS
// (256 KB of the 320 KB/WGP), preloaded ONCE by the Tensor Data Mover:
// a single tensor_load_to_lds DMA (2D tile, row stride HID) instead of a
// 256-iteration vector load/store loop. Tracked by TENSORcnt.
// State: hhat = exp(logits) (unnormalized softmax), S = sum(hhat).
// Next step uses (hhat @ Wh)/S_prev; y = Y/S + fc_b.
// slots: [ S0, S1, Y0, Y1 ] ping-pong on t&1. 2 grid barriers per step.
// ---------------------------------------------------------------------------
__global__ void __launch_bounds__(256) scan_kernel(
    const float* __restrict__ hxall,   // [T_LEN, HID]
    const float* __restrict__ Wh,      // [HID, HID]
    const float* __restrict__ bvec,    // [HID]
    const float* __restrict__ fcw,     // [HID]
    const float* __restrict__ fcb,     // [1]
    float* __restrict__ out,           // [T_LEN]
    float* hhat, float* partial, float* slots, unsigned* bar)
{
    extern __shared__ float whs[];               // [256][256] slab of Wh
    __shared__ float sh[256];
    __shared__ float redS[256];
    __shared__ float redY[256];

    const int tid  = threadIdx.x;
    const int g    = blockIdx.x * 256 + tid;
    const int part = blockIdx.x >> 3;            // Wh row-slab  [part*256, +256)
    const int base = part * 256;
    const int col0 = (blockIdx.x & 7) << 8;      // Wh column block
    const int outc = col0 + tid;                 // this thread's GEMV column

    unsigned* arrive = bar;
    unsigned* gen    = bar + 1;

    // ---- One-time slab preload via Tensor Data Mover (wave 0 issues the DMA;
    //      TDM ignores EXEC, so gate to a single wave to get one descriptor).
    //      D#: 2D tile 256x256 x 4B from Wh[base][col0], row stride HID.
    if (tid < 32) {
        unsigned long long ga =
            (unsigned long long)(const void*)(Wh + (size_t)base * HID + col0);
        unsigned ldsaddr = (unsigned)(uintptr_t)(void*)whs;   // low 32b = LDS offset
        u32x4 g0;
        g0[0] = 1u;                                // count=1 (valid), user mode
        g0[1] = ldsaddr;                           // lds_addr [63:32]
        g0[2] = (unsigned)ga;                      // global_addr [95:64]
        g0[3] = (unsigned)(ga >> 32) | (2u << 30); // global_addr hi | type=2 (image)
        u32x8 g1;
        g1[0] = 2u << 16;                          // wg_mask=0, data_size=2 (4B)
        g1[1] = (HID & 0xFFFFu) << 16;             // tensor_dim0 lo16 -> bits 63:48
        g1[2] = (HID & 0xFFFFu) << 16;             // dim0 hi=0 | tensor_dim1 lo16
        g1[3] = 256u << 16;                        // dim1 hi=0 | tile_dim0=256
        g1[4] = 256u;                              // tile_dim1=256, tile_dim2=0
        g1[5] = (unsigned)HID;                     // tensor_dim0_stride lo32 = 2048
        g1[6] = 0u;                                // stride hi | dim1_stride lo
        g1[7] = 0u;
        asm volatile("tensor_load_to_lds %0, %1"
                     :: "s"(g0), "s"(g1)
                     : "memory");
        __builtin_amdgcn_s_wait_tensorcnt(0);
    }
    __syncthreads();

    for (int t = 0; t < T_LEN; ++t) {
        // ---- Phase A: partial[part][outc] = sum_r hhat[base+r] * Wh[base+r][outc]
        sh[tid] = hhat[base + tid];
        __syncthreads();
        float acc = 0.f;
#pragma unroll 8
        for (int r = 0; r < 256; ++r)
            acc = __builtin_fmaf(sh[r], whs[r * 256 + tid], acc);
        partial[part * HID + outc] = acc;

        if (g == NTHR - 1 && t > 0) {            // emit y[t-1]
            int p = (t - 1) & 1;
            out[t - 1] = slots[2 + p] / slots[p] + fcb[0];
        }
        if (g == NTHR - 2) {                     // reset this step's accumulators
            slots[t & 1] = 0.f;
            slots[2 + (t & 1)] = 0.f;
        }
        grid_barrier(arrive, gen);

        // ---- Phase B: combine partials, exp, accumulate S and Y
        float eS = 0.f, eY = 0.f;
        if (g < HID) {
            float Sprev = slots[(t + 1) & 1];    // previous step's sum (init 1.0)
            float l = hxall[(size_t)t * HID + g] + bvec[g];
            float psum = 0.f;
#pragma unroll
            for (int p = 0; p < 8; ++p) psum += partial[p * HID + g];
            l += psum / Sprev;
            float e = __expf(l);                 // shift-free: logits are O(+-6)
            hhat[g] = e;
            eS = e;
            eY = e * fcw[g];
            if (t + 1 < T_LEN)                   // pull next row toward the WGP
                __builtin_prefetch(&hxall[(size_t)(t + 1) * HID + g], 0, 3);
        }
        if (blockIdx.x < 8) {                    // first 8 WGs own g < HID
            redS[tid] = eS; redY[tid] = eY;
            __syncthreads();
            for (int s = 128; s > 0; s >>= 1) {
                if (tid < s) { redS[tid] += redS[tid + s]; redY[tid] += redY[tid + s]; }
                __syncthreads();
            }
            if (tid == 0) {
                atomicAdd(&slots[t & 1],       redS[0]);
                atomicAdd(&slots[2 + (t & 1)], redY[0]);
            }
        }
        grid_barrier(arrive, gen);
    }

    if (g == NTHR - 1) {
        int p = (T_LEN - 1) & 1;
        out[T_LEN - 1] = slots[2 + p] / slots[p] + fcb[0];
    }
}

// ---------------------------------------------------------------------------
// Kernel 0: init persistent state (deterministic per call, graph-safe).
// ---------------------------------------------------------------------------
__global__ void init_kernel(float* hhat, float* slots, unsigned* bar)
{
    int i = blockIdx.x * 256 + threadIdx.x;
    if (i < HID) hhat[i] = 0.f;                  // h0 = 0  (hhat=0 with S_prev=1)
    if (i == 0) {
        slots[0] = 0.f; slots[1] = 1.0f;         // S ping-pong; S_prev(t=0)=1
        slots[2] = 0.f; slots[3] = 0.f;          // Y ping-pong
        bar[0] = 0u; bar[1] = 0u;                // arrive, generation
    }
}

extern "C" void kernel_launch(void* const* d_in, const int* in_sizes, int n_in,
                              void* d_out, int out_size, void* d_ws, size_t ws_size,
                              hipStream_t stream)
{
    const float* x    = (const float*)d_in[0];   // [512, 8192]
    const float* Wx   = (const float*)d_in[1];   // [512, 2048]
    const float* Wh   = (const float*)d_in[2];   // [2048, 2048]
    const float* bvec = (const float*)d_in[3];   // [2048]
    const float* fcw  = (const float*)d_in[4];   // [1, 2048]
    const float* fcb  = (const float*)d_in[5];   // [1]
    float* out = (float*)d_out;                  // [8192]

    // Workspace: hxall (64 MB) | hhat | partial | slots | barrier | xT_bf16 | Wx_bf16
    float* ws      = (float*)d_ws;
    float* hxall   = ws;
    float* hhat    = hxall + (size_t)T_LEN * HID;
    float* partial = hhat + HID;
    float* slots   = partial + 8 * HID;
    unsigned* bar  = (unsigned*)(slots + 8);
    __bf16* xbf    = (__bf16*)(bar + 8);         // 16B-aligned (offset mult of 16)
    __bf16* wxbf   = xbf + (size_t)T_LEN * IN_SZ;

    init_kernel<<<8, 256, 0, stream>>>(hhat, slots, bar);
    convw_kernel<<<(IN_SZ * HID) / 256, 256, 0, stream>>>(Wx, wxbf);
    transpose_conv_kernel<<<dim3(T_LEN / 32, IN_SZ / 32), 256, 0, stream>>>(x, xbf);
    gemm_hx_kernel<<<(T_LEN / 16) * (HID / (16 * NT)) / 8, 256, 0, stream>>>(xbf, wxbf, hxall);
    scan_kernel<<<NWG, 256, 256 * 256 * sizeof(float), stream>>>(
        hxall, Wh, bvec, fcw, fcb, out, hhat, partial, slots, bar);
}
